// PowerFlowUnconstrainedGNN_12678743458344
// MI455X (gfx1250) — matrix operand order repaired and checked by reading, not compiled, not run on hardware
//
#include <hip/hip_runtime.h>
#include <hip/hip_bf16.h>
#include <stdint.h>

#define N_NODES   100000
#define N_EDGES   1600000
#define HIDDEN    128
#define EDGE_DIM  4
#define IN_DIM    2
#define N_LAYERS  3
#define MSG_IN    134          // 2 + 128 + 4
#define KCHUNKS   5            // K padded 134 -> 160 = 5 x 32
#define NTILES    8            // 128 / 16 output column tiles
#define NF_STRIDE 68           // dwords per packed node row (130 bf16 = 65 dw + 3 pad, 16B mult)
#define ROW_DW    80           // dwords per staged LDS A row (160 bf16)
#define TILE_EDGES 16
#define NUM_TILES (N_EDGES / TILE_EDGES)   // 100000
#define BPACK_WORDS (KCHUNKS * NTILES * 32 * 8)   // 10240 dwords per layer

typedef __bf16 bf16_t;
typedef bf16_t v16bf __attribute__((ext_vector_type(16)));
typedef bf16_t v8bf  __attribute__((ext_vector_type(8)));
typedef float  v8f   __attribute__((ext_vector_type(8)));

// ---- CDNA5 async global->LDS copies (ASYNCcnt), guarded: fall back to sync path ----
#if defined(__has_builtin)
#if __has_builtin(__builtin_amdgcn_global_load_async_to_lds_b128) && \
    __has_builtin(__builtin_amdgcn_global_load_async_to_lds_b32)  && \
    __has_builtin(__builtin_amdgcn_s_wait_asynccnt)
#define USE_ASYNC_LDS 1
#endif
#endif
#ifndef USE_ASYNC_LDS
#define USE_ASYNC_LDS 0
#endif

// Exact parameter types per hipcc diagnostic:
//   b128: (v4i addrspace(1)*, v4i addrspace(3)*, imm int, imm int)
//   b32 : (int addrspace(1)*, int addrspace(3)*, imm int, imm int)
typedef int v4i_ __attribute__((__vector_size__(4 * sizeof(int))));
#define G128(p) ((__attribute__((address_space(1))) v4i_*)(p))
#define L128(p) ((__attribute__((address_space(3))) v4i_*)(p))
#define G32(p)  ((__attribute__((address_space(1))) int*)(p))
#define L32(p)  ((__attribute__((address_space(3))) int*)(p))

__device__ __forceinline__ uint32_t f2bf(float f) {
    uint32_t u = __float_as_uint(f);
    return (u + 0x7FFFu + ((u >> 16) & 1u)) >> 16;   // RNE
}
__device__ __forceinline__ uint32_t packbf(float lo, float hi) {
    return f2bf(lo) | (f2bf(hi) << 16);
}

// h0 = P_Q_inj @ W_in + b_in ; V_pred = (1, 0)
__global__ void k_node_init(const float* __restrict__ pq,
                            const float* __restrict__ Win,
                            const float* __restrict__ bin,
                            float* __restrict__ h, float* __restrict__ V) {
    const int n = blockIdx.x;
    const int j = threadIdx.x;                 // 128 threads
    const float p = pq[n * 2 + 0], q = pq[n * 2 + 1];
    h[(size_t)n * HIDDEN + j] = p * Win[j] + q * Win[HIDDEN + j] + bin[j];
    if (j == 0) { V[n * 2 + 0] = 1.0f; V[n * 2 + 1] = 0.0f; }
}

// edge features fp32 [E][4] -> packed bf16 pairs [E*2] dwords
__global__ void k_ef_cvt(const float* __restrict__ ef, uint32_t* __restrict__ efb, int ndw) {
    const int t = blockIdx.x * blockDim.x + threadIdx.x;
    if (t < ndw) efb[t] = packbf(ef[2 * t], ef[2 * t + 1]);
}

// W_msg [3][134][128] fp32 -> per-lane B fragment layout, bf16 pairs, K-padded w/ zeros.
// word index = ((l*5+kk)*8 + nt)*256 + lane*8 + j ; holds (K=k0, K=k0+1) for column n
__global__ void k_wpack(const float* __restrict__ Wmsg, uint32_t* __restrict__ Bp) {
    const int idx = blockIdx.x * blockDim.x + threadIdx.x;
    if (idx >= N_LAYERS * BPACK_WORDS) return;
    const int j    = idx & 7;
    const int lane = (idx >> 3) & 31;
    const int nt   = (idx >> 8) & 7;
    const int t    = idx >> 11;          // l*5 + kk
    const int kk   = t % KCHUNKS;
    const int l    = t / KCHUNKS;
    const int n    = nt * 16 + (lane & 15);
    const int k0   = kk * 32 + ((lane >> 4) * 16) + 2 * j;
    const float lo = (k0     < MSG_IN) ? Wmsg[((size_t)l * MSG_IN + k0)     * HIDDEN + n] : 0.0f;
    const float hi = (k0 + 1 < MSG_IN) ? Wmsg[((size_t)l * MSG_IN + k0 + 1) * HIDDEN + n] : 0.0f;
    Bp[idx] = packbf(lo, hi);
}

__global__ void k_zero(uint4* __restrict__ p, int n4) {
    const int t = blockIdx.x * blockDim.x + threadIdx.x;
    if (t < n4) p[t] = uint4{0u, 0u, 0u, 0u};
}

// nf row (68 dw): dw0=(V0,V1), dw1..64=(h pairs), dw65..67 = 0
__global__ void k_pack_nodes(const float* __restrict__ V, const float* __restrict__ h,
                             uint32_t* __restrict__ nf) {
    const int n = blockIdx.x, t = threadIdx.x;   // 64 threads
    uint32_t* row = nf + (size_t)n * NF_STRIDE;
    row[t + 1] = packbf(h[(size_t)n * HIDDEN + 2 * t], h[(size_t)n * HIDDEN + 2 * t + 1]);
    if (t == 0) row[0] = packbf(V[n * 2], V[n * 2 + 1]);
    if (t >= 1 && t <= 3) row[64 + t] = 0u;
}

// Edge message GEMM (bf16 WMMA) + scatter-sum to receivers.
// One wave handles a 16-edge x 128-out tile: 5 K-chunks x 8 N-tiles WMMAs.
__global__ void __launch_bounds__(128)
k_edge_msg(const uint32_t* __restrict__ nf,
           const uint32_t* __restrict__ efb,
           const uint32_t* __restrict__ Bp,      // layer-offset, BPACK_WORDS dwords
           const int* __restrict__ senders,
           const int* __restrict__ receivers,
           const float* __restrict__ bmsg,       // layer-offset, 128 floats
           float* __restrict__ agg) {
    __shared__ __align__(32) uint32_t ldsB[BPACK_WORDS];              // 40 KB
    __shared__ __align__(32) uint32_t ldsA[4 * TILE_EDGES * ROW_DW];  // 20 KB (wave-private slices)
    __shared__ int ldsR[4 * TILE_EDGES];

    const int tid  = threadIdx.x;
    const int wave = tid >> 5;
    const int lane = tid & 31;
    const int m    = lane & 15;      // edge row within tile / column within n-tile
    const int part = lane >> 4;

    // stage pre-swizzled weights for this layer into LDS (reused across tiles)
    for (int i = tid; i < BPACK_WORDS; i += 128) ldsB[i] = Bp[i];

    uint32_t* aBase = ldsA + wave * (TILE_EDGES * ROW_DW);
    uint32_t* row   = aBase + m * ROW_DW;
    // zero K in [134,160) pad once (weights there are zero; avoid NaN*0)
    for (int d = 67 + part; d < 80; d += 2) row[d] = 0u;

    __syncthreads();

    float bias[NTILES];
    #pragma unroll
    for (int t2 = 0; t2 < NTILES; ++t2) bias[t2] = bmsg[t2 * 16 + m];

    const int wavesTotal = (int)gridDim.x * 4;
    const int wid        = (int)blockIdx.x * 4 + wave;

    for (int tile = wid; tile < NUM_TILES; tile += wavesTotal) {
        const int ebase = tile * TILE_EDGES;
        const int e     = ebase + m;
        const int snd   = senders[e];

        // ---- gather sender row (bf16) into wave-private LDS staging ----
#if USE_ASYNC_LDS
        {
            const uint32_t* gsrc = nf + (size_t)snd * NF_STRIDE + part * 32;
            uint32_t*       ldst = row + part * 32;
            #define ACP128(i) __builtin_amdgcn_global_load_async_to_lds_b128( \
                G128(gsrc), L128(ldst), (i) * 16, 0)
            ACP128(0); ACP128(1); ACP128(2); ACP128(3);
            ACP128(4); ACP128(5); ACP128(6); ACP128(7);
            #undef ACP128
            if (part == 0) {
                __builtin_amdgcn_global_load_async_to_lds_b32(
                    G32(nf + (size_t)snd * NF_STRIDE + 64), L32(row + 64), 0, 0);
                ldsR[wave * TILE_EDGES + m] = receivers[e];
            } else {
                const uint32_t* gef = efb + (size_t)e * 2;
                __builtin_amdgcn_global_load_async_to_lds_b32(G32(gef), L32(row + 65), 0, 0);
                __builtin_amdgcn_global_load_async_to_lds_b32(G32(gef), L32(row + 65), 4, 0);
            }
            __builtin_amdgcn_s_wait_asynccnt(0);
        }
#else
        {
            const uint4* src = (const uint4*)(nf + (size_t)snd * NF_STRIDE);
            uint4* dst = (uint4*)row;
            #pragma unroll
            for (int i = 0; i < 8; ++i) dst[part * 8 + i] = src[part * 8 + i];
            if (part == 0) {
                row[64] = nf[(size_t)snd * NF_STRIDE + 64];
                ldsR[wave * TILE_EDGES + m] = receivers[e];
            } else {
                row[65] = efb[(size_t)e * 2 + 0];
                row[66] = efb[(size_t)e * 2 + 1];
            }
        }
#endif
        asm volatile("s_wait_dscnt 0" ::: "memory");   // staging writes -> fragment reads

        // ---- WMMA accumulate over 5 K-chunks x 8 N-tiles ----
        v8f acc[NTILES];
        const v8f vz = {0.f, 0.f, 0.f, 0.f, 0.f, 0.f, 0.f, 0.f};
        #pragma unroll
        for (int t2 = 0; t2 < NTILES; ++t2) acc[t2] = vz;

        #pragma unroll
        for (int kk = 0; kk < KCHUNKS; ++kk) {
            // A 16x32 bf16: lane row m; VGPR0-7 <- K=part*16..+15 (contiguous, matches B pack)
            v8bf alo = *(const v8bf*)(row + kk * 16 + part * 8);
            v8bf ahi = *(const v8bf*)(row + kk * 16 + part * 8 + 4);
            v16bf afrag = __builtin_shufflevector(alo, ahi,
                0, 1, 2, 3, 4, 5, 6, 7, 8, 9, 10, 11, 12, 13, 14, 15);
            #pragma unroll
            for (int t2 = 0; t2 < NTILES; ++t2) {
                v16bf bfrag = *(const v16bf*)(ldsB + (kk * NTILES + t2) * 256 + lane * 8);
                acc[t2] = __builtin_amdgcn_wmma_f32_16x16x32_bf16(
                    false, afrag, false, bfrag, (short)0, acc[t2], false, false);
            }
        }

        // ---- scatter-sum: D lane holds col n=t2*16+m, rows M=part*8+r ----
        int rcv[8];
        #pragma unroll
        for (int r = 0; r < 8; ++r) rcv[r] = ldsR[wave * TILE_EDGES + part * 8 + r];
        #pragma unroll
        for (int t2 = 0; t2 < NTILES; ++t2) {
            const int ncol = t2 * 16 + m;
            #pragma unroll
            for (int r = 0; r < 8; ++r) {
                unsafeAtomicAdd(agg + (size_t)rcv[r] * HIDDEN + ncol, acc[t2][r] + bias[t2]);
            }
        }
        asm volatile("s_wait_dscnt 0" ::: "memory");   // WAR vs next tile's staging writes
    }
}

// h = relu(agg); V += h @ W_out[l] + b_out[l]   (one wave per node)
__global__ void k_node_update(float* __restrict__ h, const float* __restrict__ agg,
                              const float* __restrict__ Wout, const float* __restrict__ bout,
                              float* __restrict__ V) {
    const int node = blockIdx.x * 4 + (threadIdx.x >> 5);
    const int lane = threadIdx.x & 31;
    float p0 = 0.f, p1 = 0.f;
    #pragma unroll
    for (int q2 = 0; q2 < 4; ++q2) {
        const int j = lane + q2 * 32;
        float a = agg[(size_t)node * HIDDEN + j];
        a = a > 0.f ? a : 0.f;
        h[(size_t)node * HIDDEN + j] = a;
        p0 += a * Wout[j * 2 + 0];
        p1 += a * Wout[j * 2 + 1];
    }
    #pragma unroll
    for (int ofs = 16; ofs > 0; ofs >>= 1) {
        p0 += __shfl_down(p0, ofs, 32);
        p1 += __shfl_down(p1, ofs, 32);
    }
    if (lane == 0) {
        V[node * 2 + 0] += p0 + bout[0];
        V[node * 2 + 1] += p1 + bout[1];
    }
}

extern "C" void kernel_launch(void* const* d_in, const int* in_sizes, int n_in,
                              void* d_out, int out_size, void* d_ws, size_t ws_size,
                              hipStream_t stream) {
    (void)in_sizes; (void)n_in; (void)out_size; (void)ws_size;
    const float* pq        = (const float*)d_in[0];
    const int*   senders   = (const int*)  d_in[1];
    const int*   receivers = (const int*)  d_in[2];
    const float* ef        = (const float*)d_in[3];
    const float* Win       = (const float*)d_in[4];
    const float* bin       = (const float*)d_in[5];
    const float* Wmsg      = (const float*)d_in[6];
    const float* bmsg      = (const float*)d_in[7];
    const float* Wout      = (const float*)d_in[8];
    const float* bout      = (const float*)d_in[9];
    float* V = (float*)d_out;   // V_pred lives in d_out across all layers

    char* ws = (char*)d_ws;
    size_t off = 0;
    auto alloc = [&](size_t bytes) -> void* {
        off = (off + 255) & ~(size_t)255;
        void* p = ws + off;
        off += bytes;
        return p;
    };
    float*    h   = (float*)   alloc((size_t)N_NODES * HIDDEN * sizeof(float));     // 51.2 MB
    float*    agg = (float*)   alloc((size_t)N_NODES * HIDDEN * sizeof(float));     // 51.2 MB
    uint32_t* nf  = (uint32_t*)alloc((size_t)N_NODES * NF_STRIDE * 4);              // 27.2 MB
    uint32_t* efb = (uint32_t*)alloc((size_t)N_EDGES * 2 * 4);                      // 12.8 MB
    uint32_t* Bp  = (uint32_t*)alloc((size_t)N_LAYERS * BPACK_WORDS * 4);           // 123 KB

    k_node_init<<<N_NODES, HIDDEN, 0, stream>>>(pq, Win, bin, h, V);
    {
        const int ndw = N_EDGES * 2;
        k_ef_cvt<<<(ndw + 255) / 256, 256, 0, stream>>>(ef, efb, ndw);
    }
    k_wpack<<<(N_LAYERS * BPACK_WORDS + 255) / 256, 256, 0, stream>>>(Wmsg, Bp);

    for (int l = 0; l < N_LAYERS; ++l) {
        const int n4 = N_NODES * HIDDEN / 4;
        k_zero<<<(n4 + 255) / 256, 256, 0, stream>>>((uint4*)agg, n4);
        k_pack_nodes<<<N_NODES, 64, 0, stream>>>(V, h, nf);
        k_edge_msg<<<2048, 128, 0, stream>>>(nf, efb, Bp + (size_t)l * BPACK_WORDS,
                                             senders, receivers, bmsg + (size_t)l * HIDDEN, agg);
        k_node_update<<<N_NODES / 4, 128, 0, stream>>>(h, agg,
                                                       Wout + (size_t)l * HIDDEN * IN_DIM,
                                                       bout + (size_t)l * IN_DIM, V);
    }
}
